// IQNRNNDistAgent_69166153335459
// MI455X (gfx1250) — matrix operand order repaired and checked by reading, not compiled, not run on hardware
//
#include <hip/hip_runtime.h>
#include <math.h>

typedef _Float16 h16;
typedef __attribute__((ext_vector_type(16))) _Float16 v16h;
typedef __attribute__((ext_vector_type(8)))  _Float16 v8h;
typedef __attribute__((ext_vector_type(8)))  float    v8f;

#define BS_   4096
#define E_    256
#define H_    128
#define NQ_   8
#define QED_  64
#define NA_   14

// ---- Tensor Data Mover availability (clang-23 / therock 6-arg builtin) ---
#if defined(__AMDGCN__) && __has_builtin(__builtin_amdgcn_tensor_load_to_lds) && \
    __has_builtin(__builtin_amdgcn_s_wait_tensorcnt)
#define USE_TDM 1
typedef __attribute__((ext_vector_type(4))) unsigned int u32x4;
typedef __attribute__((ext_vector_type(8))) int          i32x8;
typedef __attribute__((ext_vector_type(4))) int          i32x4;
#else
#define USE_TDM 0
#endif

// ---------------- WMMA helpers (CDNA5 16x16x32 f16 -> f32) ----------------

__device__ __forceinline__ v8f wmma_f16(v16h a, v16h b, v8f c) {
  // (neg_a, A, neg_b, B, c_mod, C, reuse_a, reuse_b)
  return __builtin_amdgcn_wmma_f32_16x16x32_f16(false, a, false, b, (short)0, c,
                                                false, false);
}

// A-fragment (16x32, row-major in LDS with stride ld):
// lane m=lane&15, half hs=lane>>4: elems 0..7 -> K = hs*8+j ; elems 8..15 -> K = 16+hs*8+j
__device__ __forceinline__ v16h ldsA(const h16* base, int ld, int k0, int lane) {
  const int m  = lane & 15;
  const int hs = (lane >> 4) & 1;
  const h16* p = base + m * ld + k0 + hs * 8;
  v8h lo = *(const v8h*)(p);
  v8h hi = *(const v8h*)(p + 16);
  v16h a;
#pragma unroll
  for (int j = 0; j < 8; ++j) { a[j] = lo[j]; a[j + 8] = hi[j]; }
  return a;
}

// B-fragment (32x16) from a transposed weight Bt[n][k] (rows contiguous in K):
// lane n=lane&15, half hs=lane>>4: elems 0..7 -> K = hs*8+j ; elems 8..15 -> K = 16+hs*8+j, column n0+n
__device__ __forceinline__ v16h gblBT(const h16* bt, int ld, int n0, int k0, int lane) {
  const int n  = lane & 15;
  const int hs = (lane >> 4) & 1;
  const h16* p = bt + (size_t)(n0 + n) * ld + k0 + hs * 8;
  v8h lo = *(const v8h*)(p);
  v8h hi = *(const v8h*)(p + 16);
  v16h b;
#pragma unroll
  for (int j = 0; j < 8; ++j) { b[j] = lo[j]; b[j + 8] = hi[j]; }
  return b;
}

#if USE_TDM
// Issue one TDM 2D tile load: rows x width f32 elements, row stride in elts.
// D# group0: count=1, lds_addr, 57-bit global_addr, type=2 ("image").
// D# group1: data_size=4B, tensor_dim0/1 == tile_dim0/1 (no OOB), dim0 stride.
// Groups 2/3 zero (<=2D tensor).
__device__ __forceinline__ void tdm_load_2d(const void* gsrc, unsigned lds_addr,
                                            unsigned width_elts, unsigned rows,
                                            unsigned stride_elts) {
  unsigned long long ga = (unsigned long long)(uintptr_t)gsrc;
  u32x4 g0;
  g0[0] = 1u;                                           // count = 1 (valid D#)
  g0[1] = lds_addr;                                     // LDS byte address
  g0[2] = (unsigned)(ga & 0xFFFFFFFFu);                 // global_addr[31:0]
  g0[3] = (unsigned)((ga >> 32) & 0x01FFFFFFu) | (2u << 30); // [56:32] | type=2
  i32x8 g1;
  g1[0] = (int)(2u << 16);                              // data_size = 4 bytes
  g1[1] = (int)((width_elts & 0xFFFFu) << 16);          // tensor_dim0[15:0]
  g1[2] = (int)((width_elts >> 16) | ((rows & 0xFFFFu) << 16)); // dim0 hi | dim1 lo
  g1[3] = (int)((rows >> 16) | ((width_elts & 0xFFFFu) << 16)); // dim1 hi | tile_dim0
  g1[4] = (int)(rows & 0xFFFFu);                        // tile_dim1 ; tile_dim2 = 0
  g1[5] = (int)stride_elts;                             // tensor_dim0_stride[31:0]
  g1[6] = 0;                                            // stride hi | dim1_stride lo
  g1[7] = 0;
  i32x4 z4 = {0, 0, 0, 0};
  i32x8 z8 = {0, 0, 0, 0, 0, 0, 0, 0};
  // 6-arg form (clang-23 / therock headers): (g0, g1, g2, g3, g4, cpol)
  __builtin_amdgcn_tensor_load_to_lds(g0, g1, z4, z4, z8, 0);
}
#endif

// ---------------- Stage 0: hypernet hidden (prev = ones -> shared row) ----

__global__ void hh_kernel(const float* __restrict__ hyp_w1,
                          const float* __restrict__ hyp_b1,
                          float* __restrict__ hh) {
  int j = threadIdx.x;
  if (j < 64) {
    float s = hyp_b1[j];
#pragma unroll
    for (int k = 0; k < 8; ++k) s += hyp_w1[k * 64 + j];
    hh[j] = fmaxf(s, 0.0f);
  }
}

// ---------------- Stage 1: weight prep (f32 -> f16 transposed copies) -----

__global__ void prep_kernel(const float* __restrict__ fc1_w,
                            const float* __restrict__ hyp_w2,
                            const float* __restrict__ hyp_b2,
                            const float* __restrict__ gru_wi,
                            const float* __restrict__ gru_wh,
                            const float* __restrict__ phi_w,
                            const float* __restrict__ fc2_w,
                            const float* __restrict__ rq,
                            const float* __restrict__ hh,
                            h16* __restrict__ fc1_wT, h16* __restrict__ hypWT,
                            h16* __restrict__ gruWiT, h16* __restrict__ gruWhT,
                            h16* __restrict__ phiWT,  h16* __restrict__ fc2WT,
                            float* __restrict__ outRq) {
  const int NT = 178176;
  for (int t = blockIdx.x * blockDim.x + threadIdx.x; t < NT;
       t += gridDim.x * blockDim.x) {
    if (t < 32768) {                       // fc1_wT[n=0..127][k=0..255]
      int n = t >> 8, k = t & 255;
      fc1_wT[n * 256 + k] = (h16)fc1_w[k * 128 + n];
    } else if (t < 65536) {                // hypWT[o=0..255][i=0..127]
      int u = t - 32768;
      int o = u >> 7, i = u & 127;
      float s = hyp_b2[i * 256 + o];
#pragma unroll 8
      for (int j = 0; j < 64; ++j)
        s = fmaf(hh[j], hyp_w2[(size_t)j * 32768 + i * 256 + o], s);
      hypWT[o * 128 + i] = (h16)s;
    } else if (t < 114688) {               // gru_wi already [n=384][k=128]
      int u = t - 65536;
      gruWiT[u] = (h16)gru_wi[u];
    } else if (t < 163840) {
      int u = t - 114688;
      gruWhT[u] = (h16)gru_wh[u];
    } else if (t < 172032) {               // phiWT[n=0..127][k=0..63]
      int u = t - 163840;
      int n = u >> 6, k = u & 63;
      phiWT[n * 64 + k] = (h16)phi_w[k * 128 + n];
    } else if (t < 174080) {               // fc2WT[n=0..15 (pad)][k=0..127]
      int u = t - 172032;
      int n = u >> 7, k = u & 127;
      fc2WT[n * 128 + k] = (n < NA_) ? (h16)fc2_w[k * NA_ + n] : (h16)0.0f;
    } else {                               // rq passthrough output
      int u = t - 174080;
      outRq[u] = rq[u];
    }
  }
}

// ---------------- Stage 2: fused per-16-row pipeline ----------------------

__global__ __launch_bounds__(32) void fused_kernel(
    const float* __restrict__ inputs, const float* __restrict__ hidden,
    const float* __restrict__ rq,     const float* __restrict__ fc1_b,
    const float* __restrict__ merger, const float* __restrict__ gru_bi,
    const float* __restrict__ gru_bh, const float* __restrict__ phi_b,
    const float* __restrict__ fc2_b,
    const h16* __restrict__ fc1_wT, const h16* __restrict__ hypWT,
    const h16* __restrict__ gruWiT, const h16* __restrict__ gruWhT,
    const h16* __restrict__ phiWT,  const h16* __restrict__ fc2WT,
    float* __restrict__ outSorted,  float* __restrict__ outH) {

  __shared__ float stageIn[16 * E_];    // TDM landing pad (f32 tile)
  __shared__ float stageH[16 * H_];     // TDM landing pad (f32 hidden)
  __shared__ h16 aIn[16][E_ + 8];       // fc1 activations (f16)
  __shared__ h16 x1[16][H_ + 8];        // relu(fc1)
  __shared__ h16 x2[16][H_ + 8];        // merged hyper output
  __shared__ h16 hprev[16][H_ + 8];     // previous hidden (f16)
  __shared__ h16 hnew[16][H_ + 8];      // GRU output (f16)
  __shared__ h16 cosT[16][QED_ + 8];    // cos features
  __shared__ h16 qphi[16][H_ + 8];      // quantile embedding
  __shared__ float qbuf[128][16];       // final q values (pre-sort)

  const int lane = threadIdx.x & 31;
  const int n    = lane & 15;
  const int hs   = lane >> 4;
  const int row0 = blockIdx.x * 16;
  const v8f Z = {0.f, 0.f, 0.f, 0.f, 0.f, 0.f, 0.f, 0.f};

  // ---- stage inputs [16,256] and hidden [16,128] into LDS ----
#if USE_TDM
  tdm_load_2d(inputs + (size_t)row0 * E_, (unsigned)(uintptr_t)&stageIn[0],
              E_, 16, E_);
  tdm_load_2d(hidden + (size_t)row0 * H_, (unsigned)(uintptr_t)&stageH[0],
              H_, 16, H_);
  if (row0 + 16 < BS_)
    __builtin_prefetch(inputs + (size_t)(row0 + 16) * E_, 0, 0);
  __builtin_amdgcn_s_wait_tensorcnt(0);
  __syncthreads();
  for (int idx = lane; idx < 16 * E_; idx += 32)
    aIn[idx >> 8][idx & 255] = (h16)stageIn[idx];
  for (int idx = lane; idx < 16 * H_; idx += 32)
    hprev[idx >> 7][idx & 127] = (h16)stageH[idx];
#else
  {
    const float4* ip = (const float4*)(inputs + (size_t)row0 * E_);
    for (int idx = lane; idx < 16 * 64; idx += 32) {
      int r = idx >> 6, c4 = idx & 63;
      float4 v = ip[r * 64 + c4];
      int c = c4 * 4;
      aIn[r][c] = (h16)v.x; aIn[r][c + 1] = (h16)v.y;
      aIn[r][c + 2] = (h16)v.z; aIn[r][c + 3] = (h16)v.w;
    }
    const float4* hp = (const float4*)(hidden + (size_t)row0 * H_);
    for (int idx = lane; idx < 16 * 32; idx += 32) {
      int r = idx >> 5, c4 = idx & 31;
      float4 v = hp[r * 32 + c4];
      int c = c4 * 4;
      hprev[r][c] = (h16)v.x; hprev[r][c + 1] = (h16)v.y;
      hprev[r][c + 2] = (h16)v.z; hprev[r][c + 3] = (h16)v.w;
    }
    if (row0 + 16 < BS_)
      __builtin_prefetch(inputs + (size_t)(row0 + 16) * E_, 0, 0);
  }
#endif
  __syncthreads();

  // ---- fc1: x1 = relu(aIn[16,256] @ fc1_w[256,128] + b) ----
  for (int nt = 0; nt < 8; ++nt) {
    v8f acc = Z;
#pragma unroll
    for (int k0 = 0; k0 < E_; k0 += 32)
      acc = wmma_f16(ldsA(&aIn[0][0], E_ + 8, k0, lane),
                     gblBT(fc1_wT, E_, nt * 16, k0, lane), acc);
    const int col = nt * 16 + n;
    const float b = fc1_b[col];
#pragma unroll
    for (int r = 0; r < 8; ++r)
      x1[hs * 8 + r][col] = (h16)fmaxf(acc[r] + b, 0.0f);
  }
  __syncthreads();

  // ---- hypernet GEMM + merger: x2 = relu(sum_head softmax(mw)*emb) ----
  // emb = x1[16,128] @ W[128,256] ; head0 = cols 0..127, head1 = cols 128..255
  for (int nt = 0; nt < 8; ++nt) {
    v8f a0 = Z, a1 = Z;
#pragma unroll
    for (int k0 = 0; k0 < H_; k0 += 32) {
      v16h ax = ldsA(&x1[0][0], H_ + 8, k0, lane);
      a0 = wmma_f16(ax, gblBT(hypWT, H_, nt * 16, k0, lane), a0);
      a1 = wmma_f16(ax, gblBT(hypWT, H_, nt * 16 + 128, k0, lane), a1);
    }
    const int col = nt * 16 + n;
    const float w0 = merger[col], w1 = merger[H_ + col];
    const float mx = fmaxf(w0, w1);
    const float e0 = __expf(w0 - mx), e1 = __expf(w1 - mx);
    const float s0 = e0 / (e0 + e1), s1 = e1 / (e0 + e1);
#pragma unroll
    for (int r = 0; r < 8; ++r)
      x2[hs * 8 + r][col] = (h16)fmaxf(s0 * a0[r] + s1 * a1[r], 0.0f);
  }
  __syncthreads();

  // ---- GRU cell: gi = x2 @ wi^T, gh = hprev @ wh^T, gate math ----
  for (int nt = 0; nt < 8; ++nt) {
    v8f gir = Z, giz = Z, gin = Z, ghr = Z, ghz = Z, ghn = Z;
#pragma unroll
    for (int k0 = 0; k0 < H_; k0 += 32) {
      v16h ax = ldsA(&x2[0][0], H_ + 8, k0, lane);
      v16h ah = ldsA(&hprev[0][0], H_ + 8, k0, lane);
      gir = wmma_f16(ax, gblBT(gruWiT, H_, nt * 16, k0, lane), gir);
      giz = wmma_f16(ax, gblBT(gruWiT, H_, nt * 16 + 128, k0, lane), giz);
      gin = wmma_f16(ax, gblBT(gruWiT, H_, nt * 16 + 256, k0, lane), gin);
      ghr = wmma_f16(ah, gblBT(gruWhT, H_, nt * 16, k0, lane), ghr);
      ghz = wmma_f16(ah, gblBT(gruWhT, H_, nt * 16 + 128, k0, lane), ghz);
      ghn = wmma_f16(ah, gblBT(gruWhT, H_, nt * 16 + 256, k0, lane), ghn);
    }
    const int col = nt * 16 + n;
    const float bir = gru_bi[col], biz = gru_bi[col + 128], bin = gru_bi[col + 256];
    const float bhr = gru_bh[col], bhz = gru_bh[col + 128], bhn = gru_bh[col + 256];
#pragma unroll
    for (int r = 0; r < 8; ++r) {
      const int m = hs * 8 + r;
      float rg = 1.0f / (1.0f + __expf(-(gir[r] + bir + ghr[r] + bhr)));
      float zg = 1.0f / (1.0f + __expf(-(giz[r] + biz + ghz[r] + bhz)));
      float nn = tanhf(gin[r] + bin + rg * (ghn[r] + bhn));
      float hp = (float)hprev[m][col];
      float hv = (1.0f - zg) * nn + zg * hp;
      hnew[m][col] = (h16)hv;
      outH[(size_t)(row0 + m) * H_ + col] = hv;
    }
  }

  // ---- cos features: 2 groups x 8 quantiles; rq index = row0 + t ----
  for (int idx = lane; idx < 16 * QED_; idx += 32) {
    int t = idx >> 6, f = idx & 63;
    float rv = rq[row0 + t];
    cosT[t][f] = (h16)__cosf(3.14159265358979323846f * (float)f * rv);
  }
  __syncthreads();

  // ---- q_phi = relu(cos[16,64] @ phi_w[64,128] + b) ----
  for (int nt = 0; nt < 8; ++nt) {
    v8f acc = Z;
#pragma unroll
    for (int k0 = 0; k0 < QED_; k0 += 32)
      acc = wmma_f16(ldsA(&cosT[0][0], QED_ + 8, k0, lane),
                     gblBT(phiWT, QED_, nt * 16, k0, lane), acc);
    const int col = nt * 16 + n;
    const float b = phi_b[col];
#pragma unroll
    for (int r = 0; r < 8; ++r)
      qphi[hs * 8 + r][col] = (h16)fmaxf(acc[r] + b, 0.0f);
  }
  __syncthreads();

  // ---- final: q[(m,iq), a] = (h[m] .* qphi[lg*8+iq]) @ fc2_w + b ----
  for (int ft = 0; ft < 8; ++ft) {
    v8f acc = Z;
    const int rr = lane & 15;
    const int fr = ft * 16 + rr;        // row = m*8 + iq
    const int m  = fr >> 3;
    const int iq = fr & 7;
    const int qr = ((m >> 3) << 3) + iq; // group-local qphi row
#pragma unroll
    for (int k0 = 0; k0 < H_; k0 += 32) {
      const int kb = k0 + hs * 8;
      v16h a;
#pragma unroll
      for (int j = 0; j < 8; ++j)
        a[j] = (h16)((float)hnew[m][kb + j] * (float)qphi[qr][kb + j]);
#pragma unroll
      for (int j = 0; j < 8; ++j)
        a[j + 8] = (h16)((float)hnew[m][kb + 16 + j] * (float)qphi[qr][kb + 16 + j]);
      acc = wmma_f16(a, gblBT(fc2WT, H_, 0, k0, lane), acc);
    }
    const float bb = (n < NA_) ? fc2_b[n] : 0.0f;
#pragma unroll
    for (int r = 0; r < 8; ++r)
      qbuf[ft * 16 + hs * 8 + r][n] = acc[r] + bb;
  }
  __syncthreads();

  // ---- transpose to [b, action, quantile], sort 8, store ----
  for (int idx = lane; idx < 16 * NA_; idx += 32) {
    int m = idx / NA_, a = idx % NA_;
    float v[8];
#pragma unroll
    for (int i = 0; i < 8; ++i) v[i] = qbuf[m * 8 + i][a];
#pragma unroll
    for (int i = 1; i < 8; ++i) {       // insertion sort ascending
      float key = v[i];
      int j = i - 1;
      while (j >= 0 && v[j] > key) { v[j + 1] = v[j]; --j; }
      v[j + 1] = key;
    }
    size_t base = ((size_t)(row0 + m) * NA_ + a) * 8;
#pragma unroll
    for (int i = 0; i < 8; ++i) outSorted[base + i] = v[i];
  }
}

// ---------------- host-side launcher --------------------------------------

extern "C" void kernel_launch(void* const* d_in, const int* in_sizes, int n_in,
                              void* d_out, int out_size, void* d_ws, size_t ws_size,
                              hipStream_t stream) {
  const float* inputs = (const float*)d_in[0];
  const float* hidden = (const float*)d_in[1];
  const float* rq     = (const float*)d_in[2];
  const float* fc1_w  = (const float*)d_in[3];
  const float* fc1_b  = (const float*)d_in[4];
  const float* hyp_w1 = (const float*)d_in[5];
  const float* hyp_b1 = (const float*)d_in[6];
  const float* hyp_w2 = (const float*)d_in[7];
  const float* hyp_b2 = (const float*)d_in[8];
  const float* merger = (const float*)d_in[9];
  const float* gru_wi = (const float*)d_in[10];
  const float* gru_wh = (const float*)d_in[11];
  const float* gru_bi = (const float*)d_in[12];
  const float* gru_bh = (const float*)d_in[13];
  const float* phi_w  = (const float*)d_in[14];
  const float* phi_b  = (const float*)d_in[15];
  const float* fc2_w  = (const float*)d_in[16];
  const float* fc2_b  = (const float*)d_in[17];

  // workspace layout
  float* hh   = (float*)d_ws;                     //   64 f32
  h16* fc1_wT = (h16*)((char*)d_ws + 512);        // 32768 f16
  h16* hypWT  = fc1_wT + 32768;                   // 32768 f16
  h16* gruWiT = hypWT + 32768;                    // 49152 f16
  h16* gruWhT = gruWiT + 49152;                   // 49152 f16
  h16* phiWT  = gruWhT + 49152;                   //  8192 f16
  h16* fc2WT  = phiWT + 8192;                     //  2048 f16

  float* outSorted = (float*)d_out;               // 4096*14*8
  float* outH      = outSorted + 4096 * 14 * 8;   // 4096*128
  float* outRq     = outSorted + 983040;          // 4096

  hipLaunchKernelGGL(hh_kernel, dim3(1), dim3(64), 0, stream, hyp_w1, hyp_b1, hh);
  hipLaunchKernelGGL(prep_kernel, dim3(696), dim3(256), 0, stream,
                     fc1_w, hyp_w2, hyp_b2, gru_wi, gru_wh, phi_w, fc2_w, rq, hh,
                     fc1_wT, hypWT, gruWiT, gruWhT, phiWT, fc2WT, outRq);
  hipLaunchKernelGGL(fused_kernel, dim3(256), dim3(32), 0, stream,
                     inputs, hidden, rq, fc1_b, merger, gru_bi, gru_bh, phi_b,
                     fc2_b, fc1_wT, hypWT, gruWiT, gruWhT, phiWT, fc2WT,
                     outSorted, outH);
}
